// QuantumEntanglementLayer_41764261986825
// MI455X (gfx1250) — matrix operand order repaired and checked by reading, not compiled, not run on hardware
//
#include <hip/hip_runtime.h>
#include <hip/hip_bf16.h>

// ---------------------------------------------------------------------------
// QuantumEntanglementLayer for MI455X (gfx1250)
//   - 3 per-edge GEMMs on v_wmma_f32_16x16x32_bf16 (wave32 WMMA)
//   - per-layer weights DMA'd into LDS by the Tensor Data Mover
//     (tensor_load_to_lds + s_wait_tensorcnt), with TDM pad_enable used to
//     produce bank-conflict-friendly 144B LDS row stride
//   - tanh / LayerNorm / L2-normalize in fp32 VALU on the WMMA C/D layout
// ---------------------------------------------------------------------------

#define QEL_NEDGES   800000
#define QEL_NODEDIM  128
#define QEL_ENT      64
#define QEL_NLAYERS  3

// padded LDS row strides (in ushorts): +8 ushorts = +16B per row
#define XS_STRIDE    264    // 256 + 8
#define HS_STRIDE    72     // 64 + 8
#define W_STRIDE     72     // 64 + 8  (produced by TDM padding)

#define XS_USHORTS   (4 * 16 * XS_STRIDE)          // 16896
#define HS_USHORTS   (4 * 16 * HS_STRIDE)          // 4608
#define W_USHORTS    (384 * W_STRIDE)              // 27648 (256 W1 + 64 W2 + 64 M)
#define SMEM_BYTES   ((XS_USHORTS + HS_USHORTS + W_USHORTS) * 2)   // 98304 = 96KB

typedef __attribute__((ext_vector_type(16))) __bf16 v16bf;
typedef __attribute__((ext_vector_type(8)))  float  v8f;
typedef __attribute__((ext_vector_type(4)))  unsigned int u32x4;
typedef __attribute__((ext_vector_type(8)))  int i32x8;
typedef __attribute__((ext_vector_type(4)))  int i32x4;

// fp32 -> bf16 (round to nearest even), raw ushort
__device__ __forceinline__ unsigned short f2bf(float f) {
    unsigned int u = __float_as_uint(f);
    unsigned int r = (u + 0x7FFFu + ((u >> 16) & 1u)) >> 16;
    return (unsigned short)r;
}

// pack two fp32 -> two bf16 (round-half-up) in one v_perm_b32
__device__ __forceinline__ unsigned int pack2bf(float lo, float hi) {
    unsigned int ul = __float_as_uint(lo) + 0x8000u;
    unsigned int uh = __float_as_uint(hi) + 0x8000u;
    return __builtin_amdgcn_perm(uh, ul, 0x07060302u);
}

__device__ __forceinline__ float fast_tanh(float x) {
    float e = __expf(2.0f * x);
    return (e - 1.0f) / (e + 1.0f);
}

// A fragment (16x32 bf16) from LDS row-major [16][stride] tile.
// lanes 0-15: rows 0-15, K = {0..7,16..23}; lanes 16-31: K = {8..15,24..31}.
__device__ __forceinline__ v16bf load_a_frag(const unsigned short* row_base,
                                             int kbase, int lane) {
    const int c0 = (lane >= 16) ? 8 : 0;
    union { v16bf v; uint4 q[2]; } u;
    u.q[0] = *(const uint4*)(row_base + kbase + c0);
    u.q[1] = *(const uint4*)(row_base + kbase + c0 + 16);
    return u.v;
}

// B fragment (32x16 bf16) from padded LDS weight rows [K][W_STRIDE].
// lane l holds column n0 + l%16; lanes 0-15: K = {0..7,16..23}; lanes 16-31: K = {8..15,24..31}.
__device__ __forceinline__ v16bf load_b_frag(const unsigned short* w,
                                             int k0, int n0, int lane) {
    const int h8 = (lane >> 4) * 8;
    const unsigned short* p = w + k0 * W_STRIDE + n0 + (lane & 15);
    union { v16bf v; unsigned short s[16]; } u;
    #pragma unroll
    for (int i = 0; i < 8; i++) {
        u.s[i]     = p[(h8 + i) * W_STRIDE];
        u.s[8 + i] = p[(16 + h8 + i) * W_STRIDE];
    }
    return u.v;
}

// Issue one TDM copy: nrows x 64 bf16 rows, contiguous in global, into LDS
// with 16B padding after each row (pad_interval=4 -> 32 DWORDs, pad_amount=3
// -> 4 DWORDs), i.e. LDS row stride = 144B = W_STRIDE ushorts.
__device__ __forceinline__ void tdm_load_rows(const unsigned short* gsrc,
                                              unsigned int lds_byte_off,
                                              unsigned int nrows) {
    const unsigned long long ga = (unsigned long long)(size_t)gsrc;
    const unsigned int nelem = nrows * 16u;   // 8-byte elements (64 bf16 = 16)
    u32x4 g0;
    g0[0] = 1u;                                        // count=1, user mode
    g0[1] = lds_byte_off;                              // lds_addr
    g0[2] = (unsigned int)(ga & 0xFFFFFFFFu);          // global_addr[31:0]
    g0[3] = (unsigned int)((ga >> 32) & 0x01FFFFFFu)   // global_addr[56:32]
          | 0x80000000u;                               // type=2 ("image")
    i32x8 g1;
    g1[0] = (int)((3u << 16)      // data_size = 8B
                | (1u << 20)      // pad_enable
                | (4u << 22)      // pad_interval: 32 DWORDs
                | (3u << 25));    // pad_amount: 4 DWORDs (16B)
    g1[1] = (int)(nelem << 16);   // tensor_dim0[15:0]
    g1[2] = (int)(1u << 16);      // tensor_dim0[31:16]=0, tensor_dim1=1
    g1[3] = (int)(nelem << 16);   // tensor_dim1[31:16]=0, tile_dim0
    g1[4] = 0;                    // tile_dim1=0, tile_dim2=0
    g1[5] = (int)nelem;           // tensor_dim0_stride[31:0]
    g1[6] = 0;
    g1[7] = 0;
    const i32x4 z4 = {0, 0, 0, 0};
    const i32x8 z8 = {0, 0, 0, 0, 0, 0, 0, 0};
    // clang-23 / therock-10.0 6-arg form: (g0, g1, g2, g3, gx, cpol)
    __builtin_amdgcn_tensor_load_to_lds(g0, g1, z4, z4, z8, 0);
}

// ---------------------------------------------------------------------------
// Prep: fp32 weights -> bf16, packed per layer as
//   [W1 (256 rows) | W2 (64 rows) | Meff (64 rows)] x 64 bf16, Bell folded.
// ---------------------------------------------------------------------------
__global__ void qel_prep(const float* __restrict__ W1,
                         const float* __restrict__ W2,
                         const float* __restrict__ M,
                         unsigned short* __restrict__ wpack) {
    const int tid    = blockIdx.x * blockDim.x + threadIdx.x;
    const int stride = gridDim.x * blockDim.x;
    const int total  = QEL_NLAYERS * 384 * QEL_ENT;   // 73728
    const float s = 0.70710678118654752f;  // 1/sqrt(2)
    for (int i = tid; i < total; i += stride) {
        const int n = i & (QEL_ENT - 1);
        const int r = (i >> 6) % 384;
        const int l = (i >> 6) / 384;
        float v;
        if (r < 256) {
            v = W1[(l * 256 + r) * QEL_ENT + n];
        } else if (r < 320) {
            v = W2[(l * 64 + (r - 256)) * QEL_ENT + n];
        } else {
            const int k = r - 320;
            const float* mr = M + (l * 64 + k) * QEL_ENT;
            if (n < 4) {
                const float m0 = mr[0], m1 = mr[1], m2 = mr[2], m3 = mr[3];
                v = (n == 0) ? s * (m0 + m3)
                  : (n == 1) ? s * (m0 - m3)
                  : (n == 2) ? s * (m1 + m2)
                  :            s * (m1 - m2);
            } else {
                v = mr[n];
            }
        }
        wpack[i] = f2bf(v);
    }
}

// ---------------------------------------------------------------------------
// Main: 128 threads = 4 waves; each wave owns a 16-edge tile.
// ---------------------------------------------------------------------------
__global__ __launch_bounds__(128, 1) void qel_main(
    const float* __restrict__ nf,        // [50000,128]
    const int*   __restrict__ ei,        // [2, E]
    const float* __restrict__ b1,        // [3,64]
    const float* __restrict__ b2,        // [3,64]
    const float* __restrict__ lng,       // [3,64]
    const float* __restrict__ lnb,       // [3,64]
    const unsigned short* __restrict__ wpack,  // bf16 [3][384][64] packed
    float* __restrict__ out)             // [E,64]
{
    extern __shared__ __align__(16) unsigned short smem[];
    unsigned short* xsb = smem;                         // [4][16][XS_STRIDE]
    unsigned short* hsb = smem + XS_USHORTS;            // [4][16][HS_STRIDE]
    unsigned short* wls = smem + XS_USHORTS + HS_USHORTS; // [384][W_STRIDE]

    const int lane = threadIdx.x & 31;
    const int wv   = threadIdx.x >> 5;
    const int e0   = blockIdx.x * 64 + wv * 16;

    unsigned short* xw = xsb + wv * 16 * XS_STRIDE;
    unsigned short* hw = hsb + wv * 16 * HS_STRIDE;

    // Warm GL2 for the packed weights (global_prefetch_b8, lane-strided).
    __builtin_prefetch(wpack + lane * 2304, 0, 1);      // 3*384*64/32

    // TDM: each wave DMAs its quarter (96 rows) of layer 0's weights.
    const unsigned int wls_off = (unsigned int)(size_t)wls;
    tdm_load_rows(wpack + wv * 96 * QEL_ENT,
                  wls_off + (unsigned int)(wv * 96 * W_STRIDE) * 2u, 96u);

    // --- Stage gathered node features as bf16 into LDS (2 lanes/edge) ---
    {
        const int edge = lane >> 1;             // 0..15
        const int half = lane & 1;              // 0 = row feats, 1 = col feats
        const int nid  = ei[half * QEL_NEDGES + (e0 + edge)];
        const float4* src = (const float4*)(nf + (long)nid * QEL_NODEDIM);
        uint2* dst = (uint2*)&xw[edge * XS_STRIDE + half * 128];
        #pragma unroll
        for (int t = 0; t < 32; t++) {
            float4 g = src[t];
            uint2 p;
            p.x = pack2bf(g.x, g.y);
            p.y = pack2bf(g.z, g.w);
            dst[t] = p;
        }
    }
    __builtin_amdgcn_s_wait_tensorcnt(0);
    __syncthreads();

    const int c  = lane & 15;   // output column base (N = c + 16*nt)
    const int hi = lane >> 4;   // row half: rows r + 8*hi in C/D layout

    v8f tot[4];
    #pragma unroll
    for (int nt = 0; nt < 4; nt++)
        #pragma unroll
        for (int r = 0; r < 8; r++) tot[nt][r] = 0.0f;

    const unsigned short* xrow = &xw[(lane & 15) * XS_STRIDE];
    const unsigned short* hrow = &hw[(lane & 15) * HS_STRIDE];

    for (int l = 0; l < QEL_NLAYERS; l++) {
        const unsigned short* W1l = wls;                  // rows 0..255
        const unsigned short* W2l = wls + 256 * W_STRIDE; // rows 256..319
        const unsigned short* Ml  = wls + 320 * W_STRIDE; // rows 320..383

        // ---------------- GEMM1: [16,256] x [256,64] ----------------
        v8f acc[4];
        #pragma unroll
        for (int nt = 0; nt < 4; nt++)
            #pragma unroll
            for (int r = 0; r < 8; r++) acc[nt][r] = 0.0f;

        #pragma unroll
        for (int kt = 0; kt < 8; kt++) {
            v16bf a  = load_a_frag(xrow, kt * 32, lane);
            v16bf q0 = load_b_frag(W1l, kt * 32,  0, lane);
            v16bf q1 = load_b_frag(W1l, kt * 32, 16, lane);
            v16bf q2 = load_b_frag(W1l, kt * 32, 32, lane);
            v16bf q3 = load_b_frag(W1l, kt * 32, 48, lane);
            acc[0] = __builtin_amdgcn_wmma_f32_16x16x32_bf16(
                false, a, false, q0, (short)0, acc[0], false, false);
            acc[1] = __builtin_amdgcn_wmma_f32_16x16x32_bf16(
                false, a, false, q1, (short)0, acc[1], false, false);
            acc[2] = __builtin_amdgcn_wmma_f32_16x16x32_bf16(
                false, a, false, q2, (short)0, acc[2], false, false);
            acc[3] = __builtin_amdgcn_wmma_f32_16x16x32_bf16(
                false, a, false, q3, (short)0, acc[3], false, false);
        }

        // bias + tanh, write h1 to LDS (bf16)
        __syncthreads();
        #pragma unroll
        for (int nt = 0; nt < 4; nt++) {
            const float bias = b1[l * QEL_ENT + c + nt * 16];
            #pragma unroll
            for (int r = 0; r < 8; r++) {
                float v = fast_tanh(acc[nt][r] + bias);
                hw[(r + hi * 8) * HS_STRIDE + c + nt * 16] = f2bf(v);
            }
        }
        __syncthreads();

        // ---------------- GEMM2: [16,64] x [64,64] ----------------
        v8f acc2[4];
        #pragma unroll
        for (int nt = 0; nt < 4; nt++)
            #pragma unroll
            for (int r = 0; r < 8; r++) acc2[nt][r] = 0.0f;

        #pragma unroll
        for (int kt = 0; kt < 2; kt++) {
            v16bf a  = load_a_frag(hrow, kt * 32, lane);
            v16bf q0 = load_b_frag(W2l, kt * 32,  0, lane);
            v16bf q1 = load_b_frag(W2l, kt * 32, 16, lane);
            v16bf q2 = load_b_frag(W2l, kt * 32, 32, lane);
            v16bf q3 = load_b_frag(W2l, kt * 32, 48, lane);
            acc2[0] = __builtin_amdgcn_wmma_f32_16x16x32_bf16(
                false, a, false, q0, (short)0, acc2[0], false, false);
            acc2[1] = __builtin_amdgcn_wmma_f32_16x16x32_bf16(
                false, a, false, q1, (short)0, acc2[1], false, false);
            acc2[2] = __builtin_amdgcn_wmma_f32_16x16x32_bf16(
                false, a, false, q2, (short)0, acc2[2], false, false);
            acc2[3] = __builtin_amdgcn_wmma_f32_16x16x32_bf16(
                false, a, false, q3, (short)0, acc2[3], false, false);
        }

        // bias + LayerNorm over 64 cols (xor-butterfly, 16-lane halves)
        #pragma unroll
        for (int nt = 0; nt < 4; nt++) {
            const float bias = b2[l * QEL_ENT + c + nt * 16];
            #pragma unroll
            for (int r = 0; r < 8; r++) acc2[nt][r] += bias;
        }
        float mean[8], inv[8];
        #pragma unroll
        for (int r = 0; r < 8; r++) {
            float s = acc2[0][r] + acc2[1][r] + acc2[2][r] + acc2[3][r];
            float q = acc2[0][r] * acc2[0][r] + acc2[1][r] * acc2[1][r]
                    + acc2[2][r] * acc2[2][r] + acc2[3][r] * acc2[3][r];
            #pragma unroll
            for (int m = 1; m < 16; m <<= 1) {
                s += __shfl_xor(s, m, 32);
                q += __shfl_xor(q, m, 32);
            }
            const float mu = s * (1.0f / 64.0f);
            const float var = q * (1.0f / 64.0f) - mu * mu;
            mean[r] = mu;
            inv[r]  = rsqrtf(var + 1e-5f);
        }
        __syncthreads();
        #pragma unroll
        for (int nt = 0; nt < 4; nt++) {
            const float g = lng[l * QEL_ENT + c + nt * 16];
            const float b = lnb[l * QEL_ENT + c + nt * 16];
            #pragma unroll
            for (int r = 0; r < 8; r++) {
                float v = (acc2[nt][r] - mean[r]) * inv[r] * g + b;
                hw[(r + hi * 8) * HS_STRIDE + c + nt * 16] = f2bf(v);
            }
        }
        __syncthreads();

        // ---------------- GEMM3: total += h @ Meff ----------------
        #pragma unroll
        for (int kt = 0; kt < 2; kt++) {
            v16bf a  = load_a_frag(hrow, kt * 32, lane);
            v16bf q0 = load_b_frag(Ml, kt * 32,  0, lane);
            v16bf q1 = load_b_frag(Ml, kt * 32, 16, lane);
            v16bf q2 = load_b_frag(Ml, kt * 32, 32, lane);
            v16bf q3 = load_b_frag(Ml, kt * 32, 48, lane);
            tot[0] = __builtin_amdgcn_wmma_f32_16x16x32_bf16(
                false, a, false, q0, (short)0, tot[0], false, false);
            tot[1] = __builtin_amdgcn_wmma_f32_16x16x32_bf16(
                false, a, false, q1, (short)0, tot[1], false, false);
            tot[2] = __builtin_amdgcn_wmma_f32_16x16x32_bf16(
                false, a, false, q2, (short)0, tot[2], false, false);
            tot[3] = __builtin_amdgcn_wmma_f32_16x16x32_bf16(
                false, a, false, q3, (short)0, tot[3], false, false);
        }

        // Refill LDS weights for the next layer (after all reads complete).
        if (l + 1 < QEL_NLAYERS) {
            __syncthreads();
            tdm_load_rows(wpack + (l + 1) * 384 * QEL_ENT + wv * 96 * QEL_ENT,
                          wls_off + (unsigned int)(wv * 96 * W_STRIDE) * 2u, 96u);
            __builtin_amdgcn_s_wait_tensorcnt(0);
            __syncthreads();
        }
    }

    // --- Final per-row L2 normalization ---
    float scale[8];
    #pragma unroll
    for (int r = 0; r < 8; r++) {
        float q = tot[0][r] * tot[0][r] + tot[1][r] * tot[1][r]
                + tot[2][r] * tot[2][r] + tot[3][r] * tot[3][r];
        #pragma unroll
        for (int m = 1; m < 16; m <<= 1) q += __shfl_xor(q, m, 32);
        scale[r] = 1.0f / (sqrtf(q) + 1e-8f);
    }
    #pragma unroll
    for (int nt = 0; nt < 4; nt++)
        #pragma unroll
        for (int r = 0; r < 8; r++) {
            const int row = e0 + r + hi * 8;
            out[row * QEL_ENT + c + nt * 16] = tot[nt][r] * scale[r];
        }
}

// ---------------------------------------------------------------------------
extern "C" void kernel_launch(void* const* d_in, const int* in_sizes, int n_in,
                              void* d_out, int out_size, void* d_ws, size_t ws_size,
                              hipStream_t stream) {
    (void)in_sizes; (void)n_in; (void)out_size; (void)ws_size;
    const float* nf  = (const float*)d_in[0];
    const int*   ei  = (const int*)d_in[1];
    const float* W1  = (const float*)d_in[2];
    const float* b1  = (const float*)d_in[3];
    const float* W2  = (const float*)d_in[4];
    const float* b2  = (const float*)d_in[5];
    const float* lng = (const float*)d_in[6];
    const float* lnb = (const float*)d_in[7];
    const float* M   = (const float*)d_in[8];

    unsigned short* wpack = (unsigned short*)d_ws;   // bf16 [3][384][64]

    qel_prep<<<64, 256, 0, stream>>>(W1, W2, M, wpack);
    qel_main<<<QEL_NEDGES / 64, 128, SMEM_BYTES, stream>>>(
        nf, ei, b1, b2, lng, lnb, wpack, (float*)d_out);
}
